// HubbardHamiltonian_9079560864170
// MI455X (gfx1250) — compile-verified, matrix operand order checked by reading, not run on hardware
//
#include <hip/hip_runtime.h>
#include <hip/hip_bf16.h>
#include <stdint.h>

// ---------------------------------------------------------------------------
// Hubbard Hamiltonian term matrix, (B=2048) x (H=2048) f32 output.
//
// Per pair (i,j): with 16-bit occupation masks x (from a) and y (from b):
//   pc = popcount(x^y)
//   out = 4.0   if pc == 0
//       = -1.0  if pc == 2 and exactly one right-hop or exactly one left-hop
//       =  0.0  otherwise            (parity factor is provably always 0)
// The three pair counts (x.y, rh, lh) are bilinear over the 16 spin-orbitals,
// fused into ONE v_wmma_i32_16x16x64_iu8 per 16x16 tile via digit encoding:
//   dot = D1 + 32*D2 + 1024*D3      (each count <= 16 < 32, no carry)
// A u8 row:  [x*1 | f1*32 | f2*32 | 0]   f1 = rot(x)&~x, f2 = ~rot(x)&x
// B u8 col:  [y*1 | g1*1  | g2*32 | 0]   g1 = ~rot(y)&y, g2 = rot(y)&~y
//
// Roofline: 16 MB f32 store dominates (~0.7us @ 23.3 TB/s). Matrix cores do
// the 2048x2048x64 reduction; epilogue is branchless (~10 VALU ops/output).
// Each wave computes 1 M-tile x 4 N-tiles: 4 independent WMMA chains hide the
// IU8 WMMA->VALU hazard (8 coexec slots) and amortize A-fragment loads.
// ---------------------------------------------------------------------------

typedef __attribute__((ext_vector_type(8))) int v8i;

#define S_DIM 8   // sites
#define SP    2   // spins
#define NS    16  // S_DIM*SP spin-orbitals
#define NT    4   // N-tiles per wave

// -------------------- Phase 1: pack masks -> WMMA u8 fragments -------------
// Fragment layout per state i (64 B = 16 dwords): frag[i*16 + half*8 + w].
// 8-bit A-matrix 16x64 layout (ISA 7.12.2): lane<16 holds K bytes
// {0-7,16-23,32-39,48-55} in v0..v7; lane>=16 holds {8-15,24-31,40-47,56-63}.
// dword w (0..7): K-block = w>>1, nibble = half*2 + (w&1).
__global__ void __launch_bounds__(256)
hub_pack_kernel(const float* __restrict__ a, const float* __restrict__ b,
                const int* __restrict__ ss_ptr,
                uint32_t* __restrict__ Afrag, uint32_t* __restrict__ Bfrag,
                int* __restrict__ popA, int* __restrict__ popB,
                int NB, int NH) {
  int gid = blockIdx.x * blockDim.x + threadIdx.x;
  if (gid >= NB + NH) return;
  const bool isA = gid < NB;
  const int  i   = isA ? gid : gid - NB;
  const int  N   = isA ? NB : NH;
  const float* src = isA ? a : b;

  // Occupation mask: bit S=2s+sp set iff arr[s,i,1,sp] > arr[s,i,0,sp]
  unsigned mask = 0u;
#pragma unroll
  for (int s = 0; s < S_DIM; ++s) {
    const float4 v = *reinterpret_cast<const float4*>(src + ((size_t)s * N + i) * 4);
    mask |= ((v.z > v.x) ? 1u : 0u) << (2 * s);
    mask |= ((v.w > v.y) ? 1u : 0u) << (2 * s + 1);
  }

  const int ss = (*ss_ptr) & (NS - 1);
  const unsigned rot = ((mask >> ss) | (mask << (NS - ss))) & 0xFFFFu;
  const unsigned fA = rot & ~mask & 0xFFFFu;   // rot(m) & ~m
  const unsigned fB = ~rot & mask & 0xFFFFu;   // ~rot(m) & m

  unsigned blk[4], scl[4];
  if (isA) { blk[0] = mask; blk[1] = fA; blk[2] = fB; blk[3] = 0u;
             scl[0] = 1u;   scl[1] = 32u; scl[2] = 32u; scl[3] = 0u; }
  else     { blk[0] = mask; blk[1] = fB; blk[2] = fA; blk[3] = 0u;
             scl[0] = 1u;   scl[1] = 1u;  scl[2] = 32u; scl[3] = 0u; }

  uint32_t* frag = (isA ? Afrag : Bfrag) + (size_t)i * 16;
#pragma unroll
  for (int half = 0; half < 2; ++half) {
#pragma unroll
    for (int w = 0; w < 8; ++w) {
      const int      kb   = w >> 1;
      const int      nib  = half * 2 + (w & 1);
      const unsigned bits = (blk[kb] >> (nib * 4)) & 0xFu;
      const unsigned c    = scl[kb];
      unsigned word = 0u;
      word |= (bits & 1u) ? c         : 0u;
      word |= (bits & 2u) ? (c << 8)  : 0u;
      word |= (bits & 4u) ? (c << 16) : 0u;
      word |= (bits & 8u) ? (c << 24) : 0u;
      frag[half * 8 + w] = word;
    }
  }
  (isA ? popA : popB)[i] = __popc(mask);
}

// -------------------- Phase 2: 4 IU8 WMMAs per wave ------------------------
// Grid: x -> N-tile groups of NT, y -> M-tile groups of 8 (1 M-tile / wave).
__global__ void __launch_bounds__(256)
hub_wmma_kernel(const uint32_t* __restrict__ Afrag,
                const uint32_t* __restrict__ Bfrag,
                const int* __restrict__ popA, const int* __restrict__ popB,
                float* __restrict__ out, int H) {
  const int wave   = threadIdx.x >> 5;
  const int lane   = threadIdx.x & 31;
  const int half   = lane >> 4;
  const int l15    = lane & 15;
  const int tile_m = blockIdx.y * 8 + wave;
  const int tile_n0 = blockIdx.x * NT;

  // A: lane holds row M=l15's K bytes; B: lane holds column N=l15 (mirrored).
  const v8i afr = *reinterpret_cast<const v8i*>(
      Afrag + (uint32_t)(tile_m * 16 + l15) * 16u + half * 8);

  v8i bfr[NT];
#pragma unroll
  for (int j = 0; j < NT; ++j)
    bfr[j] = *reinterpret_cast<const v8i*>(
        Bfrag + (uint32_t)((tile_n0 + j) * 16 + l15) * 16u + half * 8);

  // 4 independent accumulator chains -> scheduler fills IU8 hazard slots.
  v8i d[NT];
  const v8i cz = 0;
#pragma unroll
  for (int j = 0; j < NT; ++j)
    d[j] = __builtin_amdgcn_wmma_i32_16x16x64_iu8(false, afr, false, bfr[j],
                                                  cz, false, false);

  // Row popcounts for this half's 8 rows: two b128 loads (32B contiguous).
  const int m_base = tile_m * 16 + half * 8;
  const int4 pA0 = *reinterpret_cast<const int4*>(popA + m_base);
  const int4 pA1 = *reinterpret_cast<const int4*>(popA + m_base + 4);
  const int pArr[8] = {pA0.x, pA0.y, pA0.z, pA0.w, pA1.x, pA1.y, pA1.z, pA1.w};

  // All indexing in 32 bits (<= 4M elements) -> adds, no 64-bit muls.
  const uint32_t obase0 = (uint32_t)m_base * (uint32_t)H + (uint32_t)(tile_n0 * 16 + l15);

#pragma unroll
  for (int j = 0; j < NT; ++j) {
    const int pB = popB[tile_n0 * 16 + j * 16 + l15];
    const uint32_t obase = obase0 + (uint32_t)(j * 16);
#pragma unroll
    for (int r = 0; r < 8; ++r) {
      const int dot = d[j][r];
      const int D1 = dot & 31;          // x . y
      const int D2 = (dot >> 5) & 31;   // right-hop count
      const int D3 = dot >> 10;         // left-hop count
      const int pc = pArr[r] + pB - 2 * D1;              // popcount(x^y)
      const bool hop = (pc == 2) & ((D2 == 1) | (D3 == 1));
      // Branchless: 4.0f and -1.0f cases are mutually exclusive -> OR bits.
      uint32_t bits = (pc == 0) ? 0x40800000u : 0u;      // 4.0f  (U_INT)
      bits |= hop ? 0xBF800000u : 0u;                    // -1.0f (-T_HOP)
      out[obase + (uint32_t)r * (uint32_t)H] = __builtin_bit_cast(float, bits);
    }
  }
}

// ---------------------------------------------------------------------------
extern "C" void kernel_launch(void* const* d_in, const int* in_sizes, int n_in,
                              void* d_out, int out_size, void* d_ws, size_t ws_size,
                              hipStream_t stream) {
  const float* a = (const float*)d_in[0];   // (8, NB, 2, 2) f32
  const float* b = (const float*)d_in[1];   // (8, NH, 2, 2) f32
  const int* site_stride = (const int*)d_in[2];

  const int NB = in_sizes[0] / (S_DIM * 2 * SP);  // 2048
  const int NH = in_sizes[1] / (S_DIM * 2 * SP);  // 2048

  // Workspace carve-up (256B aligned): Afrag | Bfrag | popA | popB
  uintptr_t p = (uintptr_t)d_ws;
  auto align256 = [](uintptr_t v) { return (v + 255) & ~(uintptr_t)255; };
  uint32_t* Afrag = (uint32_t*)align256(p);  p = (uintptr_t)(Afrag + (size_t)NB * 16);
  uint32_t* Bfrag = (uint32_t*)align256(p);  p = (uintptr_t)(Bfrag + (size_t)NH * 16);
  int*      popA  = (int*)align256(p);       p = (uintptr_t)(popA + NB);
  int*      popB  = (int*)align256(p);

  // Phase 1: pack both state sets (NB + NH threads)
  {
    const int total = NB + NH;
    const int blk = 256;
    hub_pack_kernel<<<(total + blk - 1) / blk, blk, 0, stream>>>(
        a, b, site_stride, Afrag, Bfrag, popA, popB, NB, NH);
  }

  // Phase 2: 8 waves/block (one M-tile each) x NT N-tiles per wave.
  {
    const int tilesM = NB / 16, tilesN = NH / 16;       // 128 x 128
    dim3 grid(tilesN / NT, tilesM / 8);                 // (32, 16)
    hub_wmma_kernel<<<grid, 256, 0, stream>>>(
        Afrag, Bfrag, popA, popB, (float*)d_out, NH);
  }
}
